// ModularAttributeVQA_29094108463881
// MI455X (gfx1250) — matrix operand order repaired and checked by reading, not compiled, not run on hardware
//
#include <hip/hip_runtime.h>
#include <hip/hip_bf16.h>

typedef __attribute__((ext_vector_type(16))) __bf16 v16bf;
typedef __attribute__((ext_vector_type(8)))  __bf16 v8bf;
typedef __attribute__((ext_vector_type(4)))  __bf16 v4bf;
typedef __attribute__((ext_vector_type(8)))  float  v8f;

#define BM 128
#define BN 128
#define BK 32
#define LDSW 40   // padded row stride (bf16 elems): 80B, multiple of 16B, conflict-free

// C[z] = act(A @ B[z] + bias[z]);  A: [M,K] f32 row-major, B: [K,N] f32 row-major.
// 256 threads = 8 waves in a 4(M) x 2(N) grid; each wave owns a 32x64 tile:
// 2 A-frags x 4 B-frags -> 8 v_wmma_f32_16x16x32_bf16 per K-step.
// Double-buffered LDS; next tile register-staged during WMMA compute.
__global__ __launch_bounds__(256)
void gemm_bf16wmma_bias_relu(const float* __restrict__ A, const float* __restrict__ Bg,
                             const float* __restrict__ bias, float* __restrict__ C,
                             int M, int N, int K, int do_relu,
                             long long bStrideZ, long long biasStrideZ, long long cStrideZ) {
  __shared__ __align__(16) __bf16 Alds[2][BM][LDSW];   // [buf][row m][k]
  __shared__ __align__(16) __bf16 Blds[2][BN][LDSW];   // [buf][col n][k]  (transposed!)

  const int tid   = threadIdx.x;
  const int lane  = tid & 31;
  const int wave  = tid >> 5;
  const int waveM = wave & 3;    // 0..3  -> M offset waveM*32
  const int waveN = wave >> 2;   // 0..1  -> N offset waveN*64
  const int hlf   = lane >> 4;   // half-wave select
  const int l16   = lane & 15;

  const float* Bz    = Bg   + (size_t)blockIdx.z * bStrideZ;
  const float* biasZ = bias + (size_t)blockIdx.z * biasStrideZ;
  float*       Cz    = C    + (size_t)blockIdx.z * cStrideZ;

  const int rowBase = blockIdx.x * BM;
  const int colBase = blockIdx.y * BN;

  // staging decomposition
  const int aR  = tid >> 3;          // A: base row (0..31), +32 per chunk iter
  const int aKQ = tid & 7;           // A: float4 chunk within the 32-wide K slab
  const int bN0 = (tid & 31) * 4;    // B: 4 consecutive columns
  const int bKK = tid >> 5;          // B: 4 consecutive k-rows = bKK*4..+3

  v8f acc[2][4];
  const v8f vzero = {0.f,0.f,0.f,0.f,0.f,0.f,0.f,0.f};
#pragma unroll
  for (int mt = 0; mt < 2; ++mt)
#pragma unroll
    for (int nt = 0; nt < 4; ++nt) acc[mt][nt] = vzero;

  // ---- prologue: stage tile k0=0 into buffer 0 ----
  {
#pragma unroll
    for (int i = 0; i < 4; ++i) {
      const int r = aR + i * 32;
      const float4 v = *(const float4*)(A + (size_t)(rowBase + r) * K + aKQ * 4);
      v4bf pk = { (__bf16)v.x, (__bf16)v.y, (__bf16)v.z, (__bf16)v.w };
      *(v4bf*)(&Alds[0][r][aKQ * 4]) = pk;
    }
    const float* bp = Bz + (size_t)(bKK * 4) * N + colBase + bN0;
    const float4 r0 = *(const float4*)(bp + 0 * (size_t)N);
    const float4 r1 = *(const float4*)(bp + 1 * (size_t)N);
    const float4 r2 = *(const float4*)(bp + 2 * (size_t)N);
    const float4 r3 = *(const float4*)(bp + 3 * (size_t)N);
    v4bf c0 = { (__bf16)r0.x, (__bf16)r1.x, (__bf16)r2.x, (__bf16)r3.x };
    v4bf c1 = { (__bf16)r0.y, (__bf16)r1.y, (__bf16)r2.y, (__bf16)r3.y };
    v4bf c2 = { (__bf16)r0.z, (__bf16)r1.z, (__bf16)r2.z, (__bf16)r3.z };
    v4bf c3 = { (__bf16)r0.w, (__bf16)r1.w, (__bf16)r2.w, (__bf16)r3.w };
    *(v4bf*)(&Blds[0][bN0 + 0][bKK * 4]) = c0;
    *(v4bf*)(&Blds[0][bN0 + 1][bKK * 4]) = c1;
    *(v4bf*)(&Blds[0][bN0 + 2][bKK * 4]) = c2;
    *(v4bf*)(&Blds[0][bN0 + 3][bKK * 4]) = c3;
  }
  __syncthreads();

  for (int k0 = 0; k0 < K; k0 += BK) {
    const int cur = (k0 >> 5) & 1;
    const bool hasNext = (k0 + BK) < K;

    // ---- issue next tile's global loads first (in flight during WMMA) ----
    float4 aReg[4], bReg[4];
    if (hasNext) {
      const int kn = k0 + BK;
#pragma unroll
      for (int i = 0; i < 4; ++i)
        aReg[i] = *(const float4*)(A + (size_t)(rowBase + aR + i * 32) * K + kn + aKQ * 4);
#pragma unroll
      for (int j = 0; j < 4; ++j)
        bReg[j] = *(const float4*)(Bz + (size_t)(kn + bKK * 4 + j) * N + colBase + bN0);
      if (kn + BK < K) {  // warm L2 two tiles ahead -> global_prefetch_b8
        __builtin_prefetch(A + (size_t)(rowBase + aR) * K + kn + BK + aKQ * 4, 0, 1);
        __builtin_prefetch(Bz + (size_t)(kn + BK + bKK * 4) * N + colBase + bN0, 0, 1);
      }
    }

    // ---- compute on current buffer: fragments in contiguous k order (same for A and B) ----
    v16bf afrag[2], bfrag[4];
#pragma unroll
    for (int mt = 0; mt < 2; ++mt) {
      // A 16x32: lane holds row M=l16; K = hlf*16 .. hlf*16+15 (contiguous in Alds)
      const __bf16* p = &Alds[cur][waveM * 32 + mt * 16 + l16][hlf * 16];
      v8bf lo = *(const v8bf*)p;
      v8bf hi = *(const v8bf*)(p + 8);
      afrag[mt] = __builtin_shufflevector(lo, hi, 0,1,2,3,4,5,6,7,8,9,10,11,12,13,14,15);
    }
#pragma unroll
    for (int nt = 0; nt < 4; ++nt) {
      // B 32x16: lane holds col N=l16; K = hlf*16 .. hlf*16+15 (contiguous in Blds)
      const __bf16* p = &Blds[cur][waveN * 64 + nt * 16 + l16][hlf * 16];
      v8bf lo = *(const v8bf*)p;
      v8bf hi = *(const v8bf*)(p + 8);
      bfrag[nt] = __builtin_shufflevector(lo, hi, 0,1,2,3,4,5,6,7,8,9,10,11,12,13,14,15);
    }
#pragma unroll
    for (int mt = 0; mt < 2; ++mt)
#pragma unroll
      for (int nt = 0; nt < 4; ++nt)
        acc[mt][nt] = __builtin_amdgcn_wmma_f32_16x16x32_bf16(
            false, afrag[mt], false, bfrag[nt], (short)0, acc[mt][nt], false, false);

    // ---- convert + store next tile into the other buffer ----
    if (hasNext) {
      const int nxt = cur ^ 1;
#pragma unroll
      for (int i = 0; i < 4; ++i) {
        v4bf pk = { (__bf16)aReg[i].x, (__bf16)aReg[i].y, (__bf16)aReg[i].z, (__bf16)aReg[i].w };
        *(v4bf*)(&Alds[nxt][aR + i * 32][aKQ * 4]) = pk;
      }
      v4bf c0 = { (__bf16)bReg[0].x, (__bf16)bReg[1].x, (__bf16)bReg[2].x, (__bf16)bReg[3].x };
      v4bf c1 = { (__bf16)bReg[0].y, (__bf16)bReg[1].y, (__bf16)bReg[2].y, (__bf16)bReg[3].y };
      v4bf c2 = { (__bf16)bReg[0].z, (__bf16)bReg[1].z, (__bf16)bReg[2].z, (__bf16)bReg[3].z };
      v4bf c3 = { (__bf16)bReg[0].w, (__bf16)bReg[1].w, (__bf16)bReg[2].w, (__bf16)bReg[3].w };
      *(v4bf*)(&Blds[nxt][bN0 + 0][bKK * 4]) = c0;
      *(v4bf*)(&Blds[nxt][bN0 + 1][bKK * 4]) = c1;
      *(v4bf*)(&Blds[nxt][bN0 + 2][bKK * 4]) = c2;
      *(v4bf*)(&Blds[nxt][bN0 + 3][bKK * 4]) = c3;
    }
    __syncthreads();
  }

  // ---- epilogue: bias + optional ReLU; C/D layout: lane=(n=l16), vgpr r -> m = hlf*8+r ----
#pragma unroll
  for (int nt = 0; nt < 4; ++nt) {
    const int col = colBase + waveN * 64 + nt * 16 + l16;
    const float bv = biasZ[col];
#pragma unroll
    for (int mt = 0; mt < 2; ++mt) {
#pragma unroll
      for (int r = 0; r < 8; ++r) {
        const int row = rowBase + waveM * 32 + mt * 16 + hlf * 8 + r;
        float v = acc[mt][nt][r] + bv;
        if (do_relu) v = v > 0.f ? v : 0.f;
        Cz[(size_t)row * N + col] = v;
      }
    }
  }
}

// x_dist[n, 0:1024]   = hd[n,0,:] * hd[n,1,:]
// x_dist[n,1024:2048] = hd[n,1,:] * hd[n,2,:]
__global__ __launch_bounds__(256)
void xdist_kernel(const float* __restrict__ hd, float* __restrict__ xd) {
  size_t idx = (size_t)blockIdx.x * 256 + threadIdx.x;   // over 2048*2048
  int n = (int)(idx >> 11);
  int c = (int)(idx & 2047);
  const float* base = hd + (size_t)n * 3 * 1024;
  float v;
  if (c < 1024) v = base[c] * base[1024 + c];
  else          v = base[1024 + (c - 1024)] * base[2048 + (c - 1024)];
  xd[idx] = v;
}

// Routed second layer: out[rowOff+n, 0:2] = h[eid[n]][n,:] @ W2[eid[n]] + b2[eid[n]]
// One wave (32 lanes) per sample.
__global__ __launch_bounds__(256)
void routed_head(const float* __restrict__ h, const float* __restrict__ W2,
                 const float* __restrict__ b2, const int* __restrict__ eid,
                 float* __restrict__ out, int M, int rowOff) {
  const int n    = (int)((blockIdx.x * 256 + threadIdx.x) >> 5);
  const int lane = threadIdx.x & 31;
  if (n >= M) return;
  const int e = eid[n];
  const float* hrow = h  + ((size_t)e * M + n) * 1024;
  const float* w    = W2 + (size_t)e * 1024 * 2;
  float a0 = 0.f, a1 = 0.f;
  for (int k = lane; k < 1024; k += 32) {
    const float hv = hrow[k];
    a0 += hv * w[k * 2 + 0];
    a1 += hv * w[k * 2 + 1];
  }
#pragma unroll
  for (int off = 16; off > 0; off >>= 1) {
    a0 += __shfl_xor(a0, off, 32);
    a1 += __shfl_xor(a1, off, 32);
  }
  if (lane == 0) {
    out[(size_t)(rowOff + n) * 2 + 0] = a0 + b2[e * 2 + 0];
    out[(size_t)(rowOff + n) * 2 + 1] = a1 + b2[e * 2 + 1];
  }
}

extern "C" void kernel_launch(void* const* d_in, const int* in_sizes, int n_in,
                              void* d_out, int out_size, void* d_ws, size_t ws_size,
                              hipStream_t stream) {
  const float* rnn_feats  = (const float*)d_in[0];   // (2048,2,512)   -> [2048][1024]
  const float* cube_feats = (const float*)d_in[1];   // (2048,2,4,3200)-> [16384][3200]
  const float* ego_feats  = (const float*)d_in[2];   // (2048,3,3712)  -> [6144][3712]
  const int*   rnn_eid    = (const int*)d_in[3];
  const int*   room_eid   = (const int*)d_in[4];
  const int*   dist_eid   = (const int*)d_in[5];
  const float* Wc      = (const float*)d_in[6];
  const float* bc      = (const float*)d_in[7];
  const float* Wd      = (const float*)d_in[8];
  const float* bd      = (const float*)d_in[9];
  const float* W1_rnn  = (const float*)d_in[10];
  const float* b1_rnn  = (const float*)d_in[11];
  const float* W2_rnn  = (const float*)d_in[12];
  const float* b2_rnn  = (const float*)d_in[13];
  const float* W1_room = (const float*)d_in[14];
  const float* b1_room = (const float*)d_in[15];
  const float* W2_room = (const float*)d_in[16];
  const float* b2_room = (const float*)d_in[17];
  const float* W1_dist = (const float*)d_in[18];
  const float* b1_dist = (const float*)d_in[19];
  const float* W2_dist = (const float*)d_in[20];
  const float* b2_dist = (const float*)d_in[21];
  float* out = (float*)d_out;

  float* ws = (float*)d_ws;
  size_t off = 0;
  float* hc    = ws + off; off += (size_t)16384 * 1024;      // also viewed as x_room [2048][8192]
  float* hd    = ws + off; off += (size_t)6144 * 1024;
  float* xd    = ws + off; off += (size_t)2048 * 2048;
  float* hrnn  = ws + off; off += (size_t)3 * 2048 * 1024;
  float* hroom = ws + off; off += (size_t)2 * 2048 * 1024;
  float* hdist = ws + off; off += (size_t)2 * 2048 * 1024;

  const dim3 blk(256);
  const long long ZW = 0;

  // Front-end layers
  gemm_bf16wmma_bias_relu<<<dim3(16384/BM, 1024/BN, 1), blk, 0, stream>>>(
      cube_feats, Wc, bc, hc, 16384, 1024, 3200, 1, ZW, ZW, ZW);
  gemm_bf16wmma_bias_relu<<<dim3(6144/BM, 1024/BN, 1), blk, 0, stream>>>(
      ego_feats, Wd, bd, hd, 6144, 1024, 3712, 1, ZW, ZW, ZW);

  // Expert first layers (experts batched via gridDim.z)
  gemm_bf16wmma_bias_relu<<<dim3(2048/BM, 1024/BN, 3), blk, 0, stream>>>(
      rnn_feats, W1_rnn, b1_rnn, hrnn, 2048, 1024, 1024, 1,
      (long long)1024 * 1024, 1024, (long long)2048 * 1024);
  gemm_bf16wmma_bias_relu<<<dim3(2048/BM, 1024/BN, 2), blk, 0, stream>>>(
      hc, W1_room, b1_room, hroom, 2048, 1024, 8192, 1,
      (long long)8192 * 1024, 1024, (long long)2048 * 1024);

  xdist_kernel<<<(2048u * 2048u) / 256u, blk, 0, stream>>>(hd, xd);

  gemm_bf16wmma_bias_relu<<<dim3(2048/BM, 1024/BN, 2), blk, 0, stream>>>(
      xd, W1_dist, b1_dist, hdist, 2048, 1024, 2048, 1,
      (long long)2048 * 1024, 1024, (long long)2048 * 1024);

  // Routed second layers -> out rows [0,2048), [2048,4096), [4096,6144)
  routed_head<<<2048 / 8, blk, 0, stream>>>(hrnn,  W2_rnn,  b2_rnn,  rnn_eid,  out, 2048, 0);
  routed_head<<<2048 / 8, blk, 0, stream>>>(hroom, W2_room, b2_room, room_eid, out, 2048, 2048);
  routed_head<<<2048 / 8, blk, 0, stream>>>(hdist, W2_dist, b2_dist, dist_eid, out, 2048, 4096);
}
